// GraphAttentionLayer_56573309223523
// MI455X (gfx1250) — compile-verified, matrix operand order checked by reading, not run on hardware
//
#include <hip/hip_runtime.h>
#include <hip/hip_bf16.h>

#define N_NODES 50000
#define N_EDGES 1600000
#define IN_CH 128
#define HEADS 8
#define OUT_CH 16
#define NEG_SLOPE 0.2f
#define NBLK_RED 1024

typedef __attribute__((ext_vector_type(16))) _Float16 v16h;
typedef __attribute__((ext_vector_type(8)))  float    v8f;

// ---------------- K1: convert weights to f16 ----------------
__global__ void cvt_weights_kernel(const float* __restrict__ Wlin,
                                   const float* __restrict__ Wout,
                                   _Float16* __restrict__ Wlin_h,
                                   _Float16* __restrict__ Wout_h) {
    int i = blockIdx.x * 256 + threadIdx.x;
    if (i < IN_CH * IN_CH) Wlin_h[i] = (_Float16)Wlin[i];
    if (i < OUT_CH * IN_CH * HEADS / HEADS) { } // no-op placeholder
    if (i < 16 * 128) Wout_h[i] = (_Float16)Wout[i];
}

// ---------------- K2: x_l = x @ W_lin^T via WMMA, store f16 ----------------
// One wave (32 lanes) per 16-row tile; 8 N-tiles of 16 cols; K=128 in 4 steps of 32.
__global__ __launch_bounds__(32) void gemm1_wmma_kernel(
    const float* __restrict__ x, const _Float16* __restrict__ Wh,
    _Float16* __restrict__ xl) {
    const int tile = blockIdx.x;
    const int lane = threadIdx.x;
    const int m    = lane & 15;
    const int kb   = (lane >> 4) * 8;   // K sub-group base within 16-block
    const int row  = tile * 16 + m;

    // A fragments: lane holds row m; halves 0..7 -> K = ks*32+kb+j ; 8..15 -> +16
    v16h A[4];
    const float* xrow = x + (size_t)row * IN_CH;
    #pragma unroll
    for (int ks = 0; ks < 4; ++ks) {
        const int k0 = ks * 32 + kb;
        #pragma unroll
        for (int j = 0; j < 8; ++j) A[ks][j]     = (_Float16)xrow[k0 + j];
        #pragma unroll
        for (int j = 0; j < 8; ++j) A[ks][8 + j] = (_Float16)xrow[k0 + 16 + j];
    }

    const int mbase = (lane >> 4) * 8;
    #pragma unroll
    for (int nt = 0; nt < 8; ++nt) {
        const int n = nt * 16 + m;         // lane's output column
        v8f acc = {};
        #pragma unroll
        for (int ks = 0; ks < 4; ++ks) {
            // B[k][n] = W_lin[n][k] (row-major [128][128]) -> contiguous in k
            v16h B;
            const _Float16* wrow = Wh + (size_t)n * IN_CH + ks * 32 + kb;
            #pragma unroll
            for (int j = 0; j < 8; ++j) B[j]     = wrow[j];
            #pragma unroll
            for (int j = 0; j < 8; ++j) B[8 + j] = wrow[16 + j];
            acc = __builtin_amdgcn_wmma_f32_16x16x32_f16(
                false, A[ks], false, B, (short)0, acc, false, false);
        }
        // D layout: lane col n, rows mbase+r
        #pragma unroll
        for (int r = 0; r < 8; ++r)
            xl[(size_t)(tile * 16 + mbase + r) * IN_CH + n] = (_Float16)acc[r];
    }
}

// ---------------- K3: per-node attention scores ----------------
__global__ void node_scores_kernel(const _Float16* __restrict__ xl,
                                   const float* __restrict__ att,
                                   float* __restrict__ sl, float* __restrict__ sr) {
    int idx = blockIdx.x * 256 + threadIdx.x;   // node*8 + head
    if (idx >= N_NODES * HEADS) return;
    const int n = idx >> 3, h = idx & 7;
    const _Float16* xr = xl + (size_t)n * IN_CH + h * OUT_CH;
    const float* al = att + h * (2 * OUT_CH);
    const float* ar = al + OUT_CH;
    float a = 0.f, b = 0.f;
    #pragma unroll
    for (int c = 0; c < OUT_CH; ++c) {
        float v = (float)xr[c];
        a += v * al[c];
        b += v * ar[c];
    }
    sl[idx] = a;
    sr[idx] = b;
}

__device__ __forceinline__ float lrelu(float s) {
    return s > 0.f ? s : NEG_SLOPE * s;
}

// ---------------- K4: per-block max of logits per head ----------------
__global__ void edge_max_kernel(const int* __restrict__ ei,
                                const float* __restrict__ sl,
                                const float* __restrict__ sr,
                                float* __restrict__ partMax) {
    __shared__ float lds[256 * HEADS];
    const int tid = threadIdx.x;
    float lm[HEADS];
    #pragma unroll
    for (int h = 0; h < HEADS; ++h) lm[h] = -3.4e38f;
    for (int e = blockIdx.x * 256 + tid; e < N_EDGES; e += gridDim.x * 256) {
        const int r = ei[e], c = ei[N_EDGES + e];
        #pragma unroll
        for (int h = 0; h < HEADS; ++h) {
            float s = lrelu(sl[r * HEADS + h] + sr[c * HEADS + h]);
            lm[h] = fmaxf(lm[h], s);
        }
    }
    #pragma unroll
    for (int h = 0; h < HEADS; ++h) lds[tid * HEADS + h] = lm[h];
    __syncthreads();
    for (int st = 128; st >= 1; st >>= 1) {
        if (tid < st)
            #pragma unroll
            for (int h = 0; h < HEADS; ++h)
                lds[tid * HEADS + h] = fmaxf(lds[tid * HEADS + h], lds[(tid + st) * HEADS + h]);
        __syncthreads();
    }
    if (tid < HEADS) partMax[blockIdx.x * HEADS + tid] = lds[tid];
}

// ---------------- K5: reduce block maxima -> head max ----------------
__global__ void reduce_max_kernel(const float* __restrict__ part, float* __restrict__ mout) {
    __shared__ float lds[256];
    const int t = threadIdx.x, h = t & 7, g = t >> 3;  // 32 groups x 8 heads
    float lm = -3.4e38f;
    for (int b = g; b < NBLK_RED; b += 32) lm = fmaxf(lm, part[b * HEADS + h]);
    lds[t] = lm;
    __syncthreads();
    for (int gs = 16; gs >= 1; gs >>= 1) {
        if (g < gs) lds[t] = fmaxf(lds[t], lds[(g + gs) * 8 + h]);
        __syncthreads();
    }
    if (t < HEADS) mout[t] = lds[t];
}

// ---------------- K6: per-block sum of exp(logit - max) ----------------
__global__ void edge_expsum_kernel(const int* __restrict__ ei,
                                   const float* __restrict__ sl,
                                   const float* __restrict__ sr,
                                   const float* __restrict__ mhead,
                                   float* __restrict__ partSum) {
    __shared__ float lds[256 * HEADS];
    const int tid = threadIdx.x;
    float mh[HEADS], ls[HEADS];
    #pragma unroll
    for (int h = 0; h < HEADS; ++h) { mh[h] = mhead[h]; ls[h] = 0.f; }
    for (int e = blockIdx.x * 256 + tid; e < N_EDGES; e += gridDim.x * 256) {
        const int r = ei[e], c = ei[N_EDGES + e];
        #pragma unroll
        for (int h = 0; h < HEADS; ++h) {
            float s = lrelu(sl[r * HEADS + h] + sr[c * HEADS + h]);
            ls[h] += __expf(s - mh[h]);
        }
    }
    #pragma unroll
    for (int h = 0; h < HEADS; ++h) lds[tid * HEADS + h] = ls[h];
    __syncthreads();
    for (int st = 128; st >= 1; st >>= 1) {
        if (tid < st)
            #pragma unroll
            for (int h = 0; h < HEADS; ++h)
                lds[tid * HEADS + h] += lds[(tid + st) * HEADS + h];
        __syncthreads();
    }
    if (tid < HEADS) partSum[blockIdx.x * HEADS + tid] = lds[tid];
}

// ---------------- K7: reduce block sums -> 1/Z per head ----------------
__global__ void reduce_sum_kernel(const float* __restrict__ part, float* __restrict__ invZ) {
    __shared__ float lds[256];
    const int t = threadIdx.x, h = t & 7, g = t >> 3;
    float ls = 0.f;
    for (int b = g; b < NBLK_RED; b += 32) ls += part[b * HEADS + h];
    lds[t] = ls;
    __syncthreads();
    for (int gs = 16; gs >= 1; gs >>= 1) {
        if (g < gs) lds[t] += lds[(g + gs) * 8 + h];
        __syncthreads();
    }
    if (t < HEADS) invZ[t] = 1.f / lds[t];
}

// ---------------- K8: fused alpha * x_l[col] @ W_out^T + b via WMMA ----------------
// One wave per 16-edge tile: M=16 edges, K=128, N=16 outputs. Alpha folded into A frag.
__global__ __launch_bounds__(32) void gemm2_wmma_kernel(
    const int* __restrict__ ei, const _Float16* __restrict__ xl,
    const float* __restrict__ sl, const float* __restrict__ sr,
    const _Float16* __restrict__ Wo, const float* __restrict__ bo,
    const float* __restrict__ mhead, const float* __restrict__ invZ,
    float* __restrict__ out) {
    const int tile = blockIdx.x;
    const int lane = threadIdx.x;
    const int m    = lane & 15;
    const int kb   = (lane >> 4) * 8;
    const int e    = tile * 16 + m;
    const int r    = ei[e];
    const int c    = ei[N_EDGES + e];

    // all 8 softmax weights for this lane's edge
    float alpha[HEADS];
    #pragma unroll
    for (int h = 0; h < HEADS; ++h) {
        float s = lrelu(sl[r * HEADS + h] + sr[c * HEADS + h]);
        alpha[h] = __expf(s - mhead[h]) * invZ[h];
    }

    // B fragments: B[k][n] = W_out[n][k], n = lane column (constant over tiles)
    v16h B[4];
    #pragma unroll
    for (int ks = 0; ks < 4; ++ks) {
        const _Float16* wrow = Wo + (size_t)m * IN_CH + ks * 32 + kb;
        #pragma unroll
        for (int j = 0; j < 8; ++j) B[ks][j]     = wrow[j];
        #pragma unroll
        for (int j = 0; j < 8; ++j) B[ks][8 + j] = wrow[16 + j];
    }

    const _Float16* xr = xl + (size_t)c * IN_CH;
    v8f acc = {};
    #pragma unroll
    for (int ks = 0; ks < 4; ++ks) {
        const int k0 = ks * 32 + kb;
        const int h1 = (ks * 32) >> 4;        // head of K-group [ks*32, +16)
        const int h2 = h1 + 1;                // head of K-group [ks*32+16, +16)
        const float a1 = alpha[h1], a2 = alpha[h2];
        v16h A;
        #pragma unroll
        for (int j = 0; j < 8; ++j) A[j]     = (_Float16)((float)xr[k0 + j] * a1);
        #pragma unroll
        for (int j = 0; j < 8; ++j) A[8 + j] = (_Float16)((float)xr[k0 + 16 + j] * a2);
        acc = __builtin_amdgcn_wmma_f32_16x16x32_f16(
            false, A, false, B[ks], (short)0, acc, false, false);
    }

    const float bn = bo[m];
    const int mb = (lane >> 4) * 8;
    #pragma unroll
    for (int r2 = 0; r2 < 8; ++r2)
        out[(size_t)(tile * 16 + mb + r2) * OUT_CH + m] = acc[r2] + bn;
}

extern "C" void kernel_launch(void* const* d_in, const int* in_sizes, int n_in,
                              void* d_out, int out_size, void* d_ws, size_t ws_size,
                              hipStream_t stream) {
    const float* x    = (const float*)d_in[0];
    const int*   ei   = (const int*)d_in[1];
    const float* Wlin = (const float*)d_in[2];
    const float* att  = (const float*)d_in[3];
    const float* Wout = (const float*)d_in[4];
    const float* bo   = (const float*)d_in[5];
    float* out = (float*)d_out;

    char* ws = (char*)d_ws;
    size_t off = 0;
    auto alloc = [&](size_t bytes) -> void* {
        void* p = (void*)(ws + off);
        off += (bytes + 255) & ~(size_t)255;
        return p;
    };
    _Float16* Wlin_h  = (_Float16*)alloc((size_t)IN_CH * IN_CH * 2);
    _Float16* Wout_h  = (_Float16*)alloc((size_t)OUT_CH * IN_CH * 2);
    _Float16* xl      = (_Float16*)alloc((size_t)N_NODES * IN_CH * 2);
    float*    sl      = (float*)alloc((size_t)N_NODES * HEADS * 4);
    float*    sr      = (float*)alloc((size_t)N_NODES * HEADS * 4);
    float*    partMax = (float*)alloc((size_t)NBLK_RED * HEADS * 4);
    float*    partSum = (float*)alloc((size_t)NBLK_RED * HEADS * 4);
    float*    mhead   = (float*)alloc(HEADS * 4);
    float*    invZ    = (float*)alloc(HEADS * 4);

    cvt_weights_kernel<<<(IN_CH * IN_CH + 255) / 256, 256, 0, stream>>>(
        Wlin, Wout, Wlin_h, Wout_h);

    gemm1_wmma_kernel<<<N_NODES / 16, 32, 0, stream>>>(x, Wlin_h, xl);

    node_scores_kernel<<<(N_NODES * HEADS + 255) / 256, 256, 0, stream>>>(
        xl, att, sl, sr);

    edge_max_kernel<<<NBLK_RED, 256, 0, stream>>>(ei, sl, sr, partMax);
    reduce_max_kernel<<<1, 256, 0, stream>>>(partMax, mhead);
    edge_expsum_kernel<<<NBLK_RED, 256, 0, stream>>>(ei, sl, sr, mhead, partSum);
    reduce_sum_kernel<<<1, 256, 0, stream>>>(partSum, invZ);

    gemm2_wmma_kernel<<<N_EDGES / 16, 32, 0, stream>>>(
        ei, xl, sl, sr, Wout_h, bo, mhead, invZ, out);
}
